// AttnDecoderRNN_27547920236694
// MI455X (gfx1250) — compile-verified, matrix operand order checked by reading, not run on hardware
//
#include <hip/hip_runtime.h>
#include <hip/hip_bf16.h>
#include <math.h>

// ---------------------------------------------------------------------------
// Problem constants (from reference): S=160, B=64, H=1024, E=512, V=64, T=16
// ---------------------------------------------------------------------------
#define SS 160
#define BB 64
#define HH 1024
#define EE 512
#define VV 64
#define TT 16

typedef __attribute__((ext_vector_type(16))) __bf16 v16bf;
typedef __attribute__((ext_vector_type(8)))  float  v8f;

// Load one bf16 WMMA fragment (16 elements for this lane) from an fp32 row.
// Per the ISA bf16 16x32 layout, this lane needs K offsets {kb..kb+7} and
// {kb+16..kb+23}: four aligned float4 (b128) loads, converted in-register.
__device__ __forceinline__ v16bf load_frag_bf16(const float* __restrict__ p) {
    const float4* p4 = reinterpret_cast<const float4*>(p);
    float4 x0 = p4[0];   // k+0..3
    float4 x1 = p4[1];   // k+4..7
    float4 x2 = p4[4];   // k+16..19
    float4 x3 = p4[5];   // k+20..23
    v16bf f;
    f[0]  = (__bf16)x0.x; f[1]  = (__bf16)x0.y; f[2]  = (__bf16)x0.z; f[3]  = (__bf16)x0.w;
    f[4]  = (__bf16)x1.x; f[5]  = (__bf16)x1.y; f[6]  = (__bf16)x1.z; f[7]  = (__bf16)x1.w;
    f[8]  = (__bf16)x2.x; f[9]  = (__bf16)x2.y; f[10] = (__bf16)x2.z; f[11] = (__bf16)x2.w;
    f[12] = (__bf16)x3.x; f[13] = (__bf16)x3.y; f[14] = (__bf16)x3.z; f[15] = (__bf16)x3.w;
    return f;
}

// ---------------------------------------------------------------------------
// WMMA GEMM:  C[M,N] = A[M,K] * Bt[N,K]^T + bias[N] (+ Cadd[M,N] if HAS_ADD)
// fp32 in, bf16 WMMA (v_wmma_f32_16x16x32_bf16), fp32 accumulate/out.
// One wave32 computes a 16x64 strip: one A fragment reused across NT=4
// adjacent 16x16 N tiles (4 accumulators, 4 WMMAs per K step of 32).
// Requires M % 16 == 0, N % 64 == 0, K % 32 == 0 (true for all uses here).
// ---------------------------------------------------------------------------
template <bool HAS_ADD>
__global__ void wmma_gemm_bias(const float* __restrict__ A,
                               const float* __restrict__ Bt,
                               const float* __restrict__ bias,
                               const float* __restrict__ Cadd,
                               float* __restrict__ C,
                               int M, int N, int K) {
    constexpr int NT = 4;
    int wave = (blockIdx.x * blockDim.x + threadIdx.x) >> 5;
    int lane = threadIdx.x & 31;
    int ngroups = N >> 6;                       // N / (16*NT)
    int tm = wave / ngroups;
    int tg = wave - tm * ngroups;
    if (tm * 16 >= M) return;                   // uniform per wave: EXEC stays all-1s

    const int m0 = tm << 4;
    const int n0 = tg << 6;
    const int l15   = lane & 15;
    const int khalf = (lane >> 4) ? 8 : 0;      // ISA bf16 fragment K split

    const float* arow = A + (size_t)(m0 + l15) * K;
    const float* brow[NT];
#pragma unroll
    for (int j = 0; j < NT; ++j)
        brow[j] = Bt + (size_t)(n0 + 16 * j + l15) * K;

    v8f acc[NT] = {};
    for (int k0 = 0; k0 < K; k0 += 32) {
        const int kb = k0 + khalf;
        __builtin_prefetch(arow + kb + 32, 0, 0);   // global_prefetch next K block
        v16bf a = load_frag_bf16(arow + kb);
#pragma unroll
        for (int j = 0; j < NT; ++j) {
            v16bf b = load_frag_bf16(brow[j] + kb);
            acc[j] = __builtin_amdgcn_wmma_f32_16x16x32_bf16(
                         false, a, false, b, (short)0, acc[j], false, false);
        }
    }

    // C/D layout: lane<16 -> rows m0+r ; lane>=16 -> rows m0+8+r
    const int mbase = m0 + ((lane >> 4) << 3);
#pragma unroll
    for (int j = 0; j < NT; ++j) {
        const int col = n0 + 16 * j + l15;
        const float bv = bias[col];
#pragma unroll
        for (int r = 0; r < 8; ++r) {
            const size_t idx = (size_t)(mbase + r) * N + col;
            float val = acc[j][r] + bv;
            if (HAS_ADD) val += Cadd[idx];
            C[idx] = val;
        }
    }
}

// ---------------------------------------------------------------------------
// LSTM cell elementwise: gate order (i,f,g,o); h0/c0 constant across steps.
// Writes `out` and mirrors it into combined[:, H:2H].
// ---------------------------------------------------------------------------
__global__ void lstm_cell(const float* __restrict__ gates,
                          const float* __restrict__ c0,
                          float* __restrict__ out,
                          float* __restrict__ combined) {
    int idx = blockIdx.x * blockDim.x + threadIdx.x;
    if (idx >= BB * HH) return;
    int b = idx / HH, h = idx - b * HH;
    const float* g = gates + (size_t)b * 4 * HH;
    float gi = g[h], gf = g[HH + h], gg = g[2 * HH + h], go = g[3 * HH + h];
    float si = 1.0f / (1.0f + __expf(-gi));
    float sf = 1.0f / (1.0f + __expf(-gf));
    float so = 1.0f / (1.0f + __expf(-go));
    float c  = sf * c0[idx] + si * tanhf(gg);
    float o  = so * tanhf(c);
    out[idx] = o;
    combined[(size_t)b * 2 * HH + HH + h] = o;
}

// ---------------------------------------------------------------------------
// Additive attention scores: one block per (s,b); block-reduce over H.
// score[b,s] = attn_b + sum_h tanh(enc_t[s,b,h] + dec_t[b,h]) * attn_W[h]
// masked to -1e9 where s >= lens[b].
// ---------------------------------------------------------------------------
__global__ void attn_scores(const float* __restrict__ enc_t,
                            const float* __restrict__ dec_t,
                            const float* __restrict__ attn_W,
                            const float* __restrict__ attn_b,
                            const int*   __restrict__ lens,
                            float* __restrict__ scores) {
    int sb = blockIdx.x;            // s*B + b (matches enc_t row layout)
    int s = sb / BB, b = sb - s * BB;
    const float* et = enc_t + (size_t)sb * HH;
    const float* dt = dec_t + (size_t)b * HH;
    float part = 0.0f;
    for (int h = threadIdx.x; h < HH; h += blockDim.x)
        part += tanhf(et[h] + dt[h]) * attn_W[h];
    __shared__ float red[256];
    red[threadIdx.x] = part;
    __syncthreads();
    for (int off = 128; off > 0; off >>= 1) {
        if (threadIdx.x < off) red[threadIdx.x] += red[threadIdx.x + off];
        __syncthreads();
    }
    if (threadIdx.x == 0) {
        float sc = red[0] + attn_b[0];
        if (s >= lens[b]) sc = -1e9f;
        scores[b * SS + s] = sc;
    }
}

// ---------------------------------------------------------------------------
// Masked softmax over S + context = attn @ encoder_outputs.  One block per b.
// Writes attention weights into d_out's context_total[b,t,s] slice and the
// context vector into combined[:, 0:H].
// ---------------------------------------------------------------------------
__global__ void softmax_context(const float* __restrict__ scores,
                                const float* __restrict__ enc_outputs,
                                float* __restrict__ ctx_out,    // d_out + T*B
                                float* __restrict__ combined,
                                int t) {
    int b = blockIdx.x, tid = threadIdx.x;
    __shared__ float attn[SS];
    __shared__ float red[256];
    float v = (tid < SS) ? scores[b * SS + tid] : -3.0e38f;
    red[tid] = v; __syncthreads();
    for (int off = 128; off > 0; off >>= 1) {
        if (tid < off) red[tid] = fmaxf(red[tid], red[tid + off]);
        __syncthreads();
    }
    float mx = red[0]; __syncthreads();
    float e = (tid < SS) ? __expf(v - mx) : 0.0f;
    red[tid] = e; __syncthreads();
    for (int off = 128; off > 0; off >>= 1) {
        if (tid < off) red[tid] += red[tid + off];
        __syncthreads();
    }
    float inv = 1.0f / red[0];
    if (tid < SS) {
        float a = e * inv;
        attn[tid] = a;
        ctx_out[((size_t)b * TT + t) * SS + tid] = a;
    }
    __syncthreads();
    for (int h = tid; h < HH; h += blockDim.x) {
        float acc = 0.0f;
        for (int s = 0; s < SS; ++s)
            acc += attn[s] * enc_outputs[((size_t)s * BB + b) * HH + h];
        combined[(size_t)b * 2 * HH + h] = acc;
    }
}

// ---------------------------------------------------------------------------
// Output projection + softmax + greedy decode + stats + next embedding.
// One block (V=64 threads) per batch element.
// ---------------------------------------------------------------------------
__global__ void decode_step(const float* __restrict__ combined,
                            const float* __restrict__ out_W,
                            const float* __restrict__ out_b,
                            const float* __restrict__ embed,
                            float* __restrict__ emb_next,
                            float* __restrict__ toks_out,    // d_out, [T,B]
                            float* __restrict__ negent_out,  // [B]
                            float* __restrict__ logp_out,    // [B]
                            int t) {
    int b = blockIdx.x, v = threadIdx.x;        // v in [0,64)
    const float* cb = combined + (size_t)b * 2 * HH;
    const float* w  = out_W    + (size_t)v * 2 * HH;
    float logit = out_b[v];
    for (int j = 0; j < 2 * HH; ++j) logit += cb[j] * w[j];

    __shared__ float sh[VV];
    __shared__ float sh2[VV];
    __shared__ int   shi[VV];
    sh[v] = logit; __syncthreads();
    for (int off = 32; off > 0; off >>= 1) {
        if (v < off) sh[v] = fmaxf(sh[v], sh[v + off]);
        __syncthreads();
    }
    float mx = sh[0]; __syncthreads();
    float e = __expf(logit - mx);
    sh[v] = e; __syncthreads();
    for (int off = 32; off > 0; off >>= 1) {
        if (v < off) sh[v] += sh[v + off];
        __syncthreads();
    }
    float p = e / sh[0];
    __syncthreads();

    // argmax with first-max tie-break; fused entropy partial sums
    sh[v] = p; shi[v] = v; sh2[v] = p * __logf(p + 1e-6f);
    __syncthreads();
    for (int off = 32; off > 0; off >>= 1) {
        if (v < off) {
            if (sh[v + off] > sh[v] ||
                (sh[v + off] == sh[v] && shi[v + off] < shi[v])) {
                sh[v] = sh[v + off]; shi[v] = shi[v + off];
            }
            sh2[v] += sh2[v + off];
        }
        __syncthreads();
    }
    int tok = shi[0];
    if (v == 0) {
        float lp = __logf(sh[0] + 1e-6f);       // sh[0] == p[tok]
        if (t == 0) { negent_out[b] = sh2[0];  logp_out[b] = lp; }
        else        { negent_out[b] += sh2[0]; logp_out[b] += lp; }
        toks_out[t * BB + b] = (float)tok;
    }
    for (int e2 = v; e2 < EE; e2 += blockDim.x)
        emb_next[(size_t)b * EE + e2] = embed[(size_t)tok * EE + e2];
}

// Initial embedding: broadcast <SOS> embedding to all batch rows.
__global__ void init_emb(const float* __restrict__ tok_embed,
                         float* __restrict__ emb) {
    int idx = blockIdx.x * blockDim.x + threadIdx.x;
    if (idx < BB * EE) emb[idx] = tok_embed[idx % EE];
}

// ---------------------------------------------------------------------------
extern "C" void kernel_launch(void* const* d_in, const int* in_sizes, int n_in,
                              void* d_out, int out_size, void* d_ws, size_t ws_size,
                              hipStream_t stream) {
    (void)in_sizes; (void)n_in; (void)out_size; (void)ws_size;
    const float* encoder_h0      = (const float*)d_in[0];
    const float* encoder_c0      = (const float*)d_in[1];
    const float* encoder_outputs = (const float*)d_in[2];
    const int*   encoder_lens    = (const int*)  d_in[3];
    const float* tok_embed       = (const float*)d_in[4];
    const float* embed           = (const float*)d_in[5];
    const float* W_ih            = (const float*)d_in[6];
    const float* W_hh            = (const float*)d_in[7];
    const float* b_ih            = (const float*)d_in[8];
    const float* b_hh            = (const float*)d_in[9];
    const float* out_W           = (const float*)d_in[10];
    const float* out_b           = (const float*)d_in[11];
    const float* enc_W           = (const float*)d_in[12];
    const float* enc_b           = (const float*)d_in[13];
    const float* dec_W           = (const float*)d_in[14];
    const float* dec_b           = (const float*)d_in[15];
    const float* attn_W          = (const float*)d_in[16];
    const float* attn_b          = (const float*)d_in[17];

    // Workspace layout (fp32)
    float* ws       = (float*)d_ws;
    float* enc_t    = ws;                              // S*B*H
    float* hh_const = enc_t    + (size_t)SS * BB * HH; // B*4H
    float* gates    = hh_const + (size_t)BB * 4 * HH;  // B*4H
    float* out_v    = gates    + (size_t)BB * 4 * HH;  // B*H
    float* dec_t    = out_v    + (size_t)BB * HH;      // B*H
    float* scores   = dec_t    + (size_t)BB * HH;      // B*S
    float* combined = scores   + (size_t)BB * SS;      // B*2H
    float* emb_cur  = combined + (size_t)BB * 2 * HH;  // B*E

    // d_out layout: toks[T*B] | context_total[B*T*S] | negent[B] | logp[B]
    float* o_toks   = (float*)d_out;
    float* o_ctx    = o_toks + TT * BB;
    float* o_negent = o_ctx  + (size_t)BB * TT * SS;
    float* o_logp   = o_negent + BB;

    auto gemm_blocks = [](int M, int N) {
        int waves = (M / 16) * (N / 64);               // 16x64 strip per wave
        return (waves + 7) / 8;                        // 8 waves / 256-thread block
    };

    // --- step-invariant precompute -----------------------------------------
    // enc_t[s*B+b, :] = encoder_outputs[s,b,:] @ enc_W^T + enc_b
    wmma_gemm_bias<false><<<gemm_blocks(SS * BB, HH), 256, 0, stream>>>(
        encoder_outputs, enc_W, enc_b, nullptr, enc_t, SS * BB, HH, HH);
    // hh_const = h0 @ W_hh^T + b_hh   (h0 = encoder_h0[0])
    wmma_gemm_bias<false><<<gemm_blocks(BB, 4 * HH), 256, 0, stream>>>(
        encoder_h0, W_hh, b_hh, nullptr, hh_const, BB, 4 * HH, HH);
    init_emb<<<(BB * EE + 255) / 256, 256, 0, stream>>>(tok_embed, emb_cur);

    // --- sequential greedy decode ------------------------------------------
    for (int t = 0; t < TT; ++t) {
        // gates = emb @ W_ih^T + b_ih + hh_const
        wmma_gemm_bias<true><<<gemm_blocks(BB, 4 * HH), 256, 0, stream>>>(
            emb_cur, W_ih, b_ih, hh_const, gates, BB, 4 * HH, EE);
        lstm_cell<<<(BB * HH + 255) / 256, 256, 0, stream>>>(
            gates, encoder_c0, out_v, combined);
        // dec_t = out @ dec_W^T + dec_b
        wmma_gemm_bias<false><<<gemm_blocks(BB, HH), 256, 0, stream>>>(
            out_v, dec_W, dec_b, nullptr, dec_t, BB, HH, HH);
        attn_scores<<<SS * BB, 256, 0, stream>>>(
            enc_t, dec_t, attn_W, attn_b, encoder_lens, scores);
        softmax_context<<<BB, 256, 0, stream>>>(
            scores, encoder_outputs, o_ctx, combined, t);
        decode_step<<<BB, VV, 0, stream>>>(
            combined, out_W, out_b, embed, emb_cur, o_toks, o_negent, o_logp, t);
    }
}
